// PraxisRecurrent_77695958385352
// MI455X (gfx1250) — matrix operand order repaired and checked by reading, not compiled, not run on hardware
//
#include <hip/hip_runtime.h>

// ---------------- problem constants ----------------
#define B_ 8
#define S_ 2048
#define D_ 1024
#define E_ 8
#define EPS_ 1e-5f

typedef __attribute__((ext_vector_type(16))) __bf16 v16bf;
typedef __attribute__((ext_vector_type(8)))  __bf16 v8bf;
typedef __attribute__((ext_vector_type(8)))  float  v8f;

union ABfrag { v16bf v; v8bf h[2]; };

static __device__ __forceinline__ unsigned short f2bf(float f) {
    unsigned int u = __builtin_bit_cast(unsigned int, f);
    u += 0x7fffu + ((u >> 16) & 1u);               // round-to-nearest-even
    return (unsigned short)(u >> 16);
}
static __device__ __forceinline__ float bf2f(unsigned short h) {
    unsigned int u = ((unsigned int)h) << 16;
    return __builtin_bit_cast(float, u);
}

// ---------------- 1) RMSNorm -> bf16 x ----------------
__global__ void rmsnorm_kernel(const float* __restrict__ in,
                               const float* __restrict__ nw,
                               unsigned short* __restrict__ xbf) {
    __shared__ float red[256];
    const int row = blockIdx.x;                    // 0 .. B*S-1
    const size_t base = (size_t)row * D_;
    const int tid = threadIdx.x;
    const int d0 = tid * 4;
    const float4 xv = *(const float4*)(in + base + d0);
    float ss = xv.x * xv.x + xv.y * xv.y + xv.z * xv.z + xv.w * xv.w;
    red[tid] = ss;
    __syncthreads();
    for (int st = 128; st > 0; st >>= 1) {
        if (tid < st) red[tid] += red[tid + st];
        __syncthreads();
    }
    const float scale = rsqrtf(red[0] * (1.0f / D_) + EPS_);
    ushort4 o;
    o.x = f2bf(xv.x * scale * nw[d0 + 0]);
    o.y = f2bf(xv.y * scale * nw[d0 + 1]);
    o.z = f2bf(xv.z * scale * nw[d0 + 2]);
    o.w = f2bf(xv.w * scale * nw[d0 + 3]);
    *(ushort4*)(xbf + base + d0) = o;
}

// ---------------- 2) mean-pool over S ----------------
__global__ void pool_kernel(const unsigned short* __restrict__ xbf,
                            float* __restrict__ pooled) {
    const int idx = blockIdx.x * 256 + threadIdx.x;  // 0 .. B*D-1
    const int b = idx / D_, d = idx % D_;
    const unsigned short* p = xbf + (size_t)b * S_ * D_ + d;
    float s = 0.f;
    for (int t = 0; t < S_; ++t) s += bf2f(p[(size_t)t * D_]);
    pooled[idx] = s * (1.0f / S_);
}

// ---------------- 3) router softmax (block per batch) ----------------
__global__ void router_kernel(const float* __restrict__ pooled,
                              const float* __restrict__ rw,
                              const float* __restrict__ rb,
                              float* __restrict__ probs,
                              float* __restrict__ aux) {
    __shared__ float red[256];
    __shared__ float logits[E_];
    const int b = blockIdx.x, tid = threadIdx.x;
    for (int e = 0; e < E_; ++e) {
        float s = 0.f;
        for (int d = tid; d < D_; d += 256) s += pooled[b * D_ + d] * rw[e * D_ + d];
        red[tid] = s;
        __syncthreads();
        for (int st = 128; st > 0; st >>= 1) {
            if (tid < st) red[tid] += red[tid + st];
            __syncthreads();
        }
        if (tid == 0) logits[e] = red[0] + rb[e];
        __syncthreads();
    }
    if (tid == 0) {
        float mx = -1e30f;
        for (int e = 0; e < E_; ++e) mx = fmaxf(mx, logits[e]);
        float sum = 0.f, ex[E_];
        for (int e = 0; e < E_; ++e) { ex[e] = __expf(logits[e] - mx); sum += ex[e]; }
        const float inv = 1.0f / sum;
        for (int e = 0; e < E_; ++e) probs[b * E_ + e] = ex[e] * inv;
        if (b == 0) aux[0] = 0.0f;                 // aux_loss
    }
}

// ---------------- 4) mix expert weights -> per-batch bf16 W ----------------
__global__ void mix_kernel(const float* __restrict__ whg,
                           const float* __restrict__ wout,
                           const float* __restrict__ probs,
                           unsigned short* __restrict__ whgbf,
                           unsigned short* __restrict__ woutbf) {
    __shared__ float p[B_ * E_];
    const int tid = threadIdx.x;
    if (tid < B_ * E_) p[tid] = probs[tid];
    __syncthreads();

    const size_t idx4 = (size_t)blockIdx.x * 256 + tid;    // one float4 per thread
    const size_t HG4 = (size_t)2 * D_ * D_ / 4;
    const float* src;
    unsigned short* dst;
    size_t estr, pos;
    if (idx4 < HG4) { pos = idx4 * 4; src = whg;  estr = (size_t)2 * D_ * D_; dst = whgbf; }
    else            { pos = (idx4 - HG4) * 4; src = wout; estr = (size_t)D_ * D_; dst = woutbf; }

    float ax[B_] = {}, ay[B_] = {}, az[B_] = {}, aw[B_] = {};
    #pragma unroll
    for (int e = 0; e < E_; ++e) {
        const float4 w = *(const float4*)(src + (size_t)e * estr + pos);
        #pragma unroll
        for (int b = 0; b < B_; ++b) {
            const float pb = p[b * E_ + e];
            ax[b] += pb * w.x; ay[b] += pb * w.y; az[b] += pb * w.z; aw[b] += pb * w.w;
        }
    }
    #pragma unroll
    for (int b = 0; b < B_; ++b) {
        ushort4 o;
        o.x = f2bf(ax[b]); o.y = f2bf(ay[b]); o.z = f2bf(az[b]); o.w = f2bf(aw[b]);
        *(ushort4*)(dst + (size_t)b * estr + pos) = o;
    }
}

// ---------------- 5/7) bf16 WMMA GEMM: C[s,o] = sum_d A[s,d]*W[o,d] ----------------
// block = 256 threads = 8 waves (2x4). Each wave: 32x64 output tile.
// Single A/B base pointer per lane; mt/nt tile displacements are constant
// element offsets that lower into the 24-bit instruction offset field.
// MODE 0: store bf16 (hg).  MODE 1: add residual, store f32 (out).
// Templated on MODE -> branch-free straight-line epilogue.
template <int MODE>
__global__ void __launch_bounds__(256)
gemm_bf16_kernel(const unsigned short* __restrict__ Abf,
                 const unsigned short* __restrict__ Wbf,
                 const float* __restrict__ resid,
                 unsigned short* __restrict__ outBf,
                 float* __restrict__ outF,
                 int N, size_t strideA, size_t strideW,
                 size_t strideC) {
    const int bz = blockIdx.z;

    const int wave = threadIdx.x >> 5;
    const int lane = threadIdx.x & 31;
    const int wm = wave >> 2, wn = wave & 3;
    const int m0 = blockIdx.y * 64 + wm * 32;
    const int n0 = blockIdx.x * 256 + wn * 64;
    const int laneM = lane & 15;
    const int khalf = lane >> 4;                   // 0: lanes 0-15, 1: lanes 16-31

    // one base pointer per operand per lane
    const unsigned short* pa = Abf + (size_t)bz * strideA
                             + (size_t)(m0 + laneM) * D_ + khalf * 8;
    const unsigned short* pb = Wbf + (size_t)bz * strideW
                             + (size_t)(n0 + laneM) * D_ + khalf * 8;

    v8f acc[2][4] = {};

    for (int k0 = 0; k0 < D_; k0 += 32) {
        v16bf afr[2], bfr[4];
        #pragma unroll
        for (int mt = 0; mt < 2; ++mt) {
            // 16-bit A 16x32 layout: lanes 0-15 hold K {0..7,16..23}, lanes 16-31 hold K {8..15,24..31}
            ABfrag a;
            a.h[0] = *(const v8bf*)(pa + mt * 16 * D_ + k0);
            a.h[1] = *(const v8bf*)(pa + mt * 16 * D_ + k0 + 16);
            afr[mt] = a.v;
        }
        #pragma unroll
        for (int nt = 0; nt < 4; ++nt) {
            // 16-bit B 32x16 layout: lanes 0-15 hold K {0..7,16..23}, lanes 16-31 hold K {8..15,24..31} for column N=laneM
            ABfrag bb;
            bb.h[0] = *(const v8bf*)(pb + nt * 16 * D_ + k0);
            bb.h[1] = *(const v8bf*)(pb + nt * 16 * D_ + k0 + 16);
            bfr[nt] = bb.v;
        }
        __builtin_prefetch(pa + k0 + 32, 0, 0);    // next K slice -> global_prefetch_b8
        __builtin_prefetch(pb + k0 + 32, 0, 0);
        #pragma unroll
        for (int mt = 0; mt < 2; ++mt)
            #pragma unroll
            for (int nt = 0; nt < 4; ++nt)
                acc[mt][nt] = __builtin_amdgcn_wmma_f32_16x16x32_bf16(
                    false, afr[mt], false, bfr[nt], (short)0, acc[mt][nt], false, false);
    }

    // C layout: VGPR i, lanes 0-15 -> M=i, lanes 16-31 -> M=8+i; N = laneM
    #pragma unroll
    for (int mt = 0; mt < 2; ++mt)
        #pragma unroll
        for (int nt = 0; nt < 4; ++nt) {
            const int ccol = n0 + nt * 16 + laneM;
            #pragma unroll
            for (int i = 0; i < 8; ++i) {
                const int rrow = m0 + mt * 16 + khalf * 8 + i;
                const size_t off = (size_t)bz * strideC + (size_t)rrow * N + ccol;
                const float v = acc[mt][nt][i];
                if (MODE == 0) {
                    outBf[off] = f2bf(v);
                } else {
                    outF[off] = v + resid[off];
                }
            }
        }
}

// ---------------- 6) per-channel sequential scan over S ----------------
// h_t = sigmoid(-g)*h_{t-1} + sigmoid(g)*g(hid), g(x)= x>=0 ? x+0.5 : sigmoid(x)
__global__ void scan_kernel(const unsigned short* __restrict__ hg,
                            const float* __restrict__ state,
                            unsigned short* __restrict__ hbf,
                            float* __restrict__ newstate) {
    const int idx = blockIdx.x * 256 + threadIdx.x;  // 0 .. B*D-1
    const int b = idx / D_, d = idx % D_;
    float h = state[b * D_ + d];
    const unsigned short* p = hg + (size_t)b * S_ * 2 * D_ + d;
    unsigned short* po = hbf + (size_t)b * S_ * D_ + d;
    for (int t = 0; t < S_; ++t) {
        const float hd = bf2f(p[0]);
        const float g  = bf2f(p[D_]);
        const float sg = 1.0f / (1.0f + __expf(-g));   // sigmoid(g)
        const float gf = (hd >= 0.0f) ? (hd + 0.5f) : (1.0f / (1.0f + __expf(-hd)));
        h = (1.0f - sg) * h + sg * gf;
        po[0] = f2bf(h);
        p  += 2 * D_;
        po += D_;
    }
    newstate[b * D_ + d] = h;
}

// ---------------- launch ----------------
extern "C" void kernel_launch(void* const* d_in, const int* in_sizes, int n_in,
                              void* d_out, int out_size, void* d_ws, size_t ws_size,
                              hipStream_t stream) {
    const float* inputs  = (const float*)d_in[0];   // (B,S,D)
    const float* state   = (const float*)d_in[1];   // (B,1,D)
    const float* nw      = (const float*)d_in[2];   // (D)
    const float* rw      = (const float*)d_in[3];   // (E,D)
    const float* rb      = (const float*)d_in[4];   // (E)
    const float* w_hg    = (const float*)d_in[5];   // (E,2D,D)
    const float* w_out   = (const float*)d_in[6];   // (E,D,D)

    float* out      = (float*)d_out;                           // (B,S,D)
    float* newstate = out + (size_t)B_ * S_ * D_;              // (B,1,D)
    float* aux      = newstate + (size_t)B_ * D_;              // scalar

    // workspace carving
    char* ws = (char*)d_ws;
    unsigned short* xbf    = (unsigned short*)ws;  ws += (size_t)B_ * S_ * D_ * 2;       // 32MB
    unsigned short* hbf    = (unsigned short*)ws;  ws += (size_t)B_ * S_ * D_ * 2;       // 32MB
    unsigned short* whgbf  = (unsigned short*)ws;  ws += (size_t)B_ * 2 * D_ * D_ * 2;   // 32MB
    unsigned short* woutbf = (unsigned short*)ws;  ws += (size_t)B_ * D_ * D_ * 2;       // 16MB
    unsigned short* hgbf   = (unsigned short*)ws;  ws += (size_t)B_ * S_ * 2 * D_ * 2;   // 64MB
    float* pooled          = (float*)ws;           ws += (size_t)B_ * D_ * 4;
    float* probs           = (float*)ws;           ws += (size_t)B_ * E_ * 4;

    // 1) RMSNorm -> bf16 x
    rmsnorm_kernel<<<B_ * S_, 256, 0, stream>>>(inputs, nw, xbf);
    // 2) mean pool
    pool_kernel<<<(B_ * D_) / 256, 256, 0, stream>>>(xbf, pooled);
    // 3) router softmax (+ aux_loss = 0)
    router_kernel<<<B_, 256, 0, stream>>>(pooled, rw, rb, probs, aux);
    // 4) mix expert weights -> per-batch bf16
    mix_kernel<<<(3 * D_ * D_ / 4) / 256, 256, 0, stream>>>(w_hg, w_out, probs, whgbf, woutbf);
    // 5) hg = x @ Whg^T (bf16 WMMA, fp32 acc), store bf16
    gemm_bf16_kernel<0><<<dim3((2 * D_) / 256, S_ / 64, B_), 256, 0, stream>>>(
        xbf, whgbf, nullptr, hgbf, nullptr,
        2 * D_, (size_t)S_ * D_, (size_t)2 * D_ * D_, (size_t)S_ * 2 * D_);
    // 6) gated linear recurrence along S
    scan_kernel<<<(B_ * D_) / 256, 256, 0, stream>>>(hgbf, state, hbf, newstate);
    // 7) out = h @ Wout^T + inputs (bf16 WMMA, fp32 acc), store f32
    gemm_bf16_kernel<1><<<dim3(D_ / 256, S_ / 64, B_), 256, 0, stream>>>(
        hbf, woutbf, inputs, nullptr, out,
        D_, (size_t)S_ * D_, (size_t)D_ * D_, (size_t)S_ * D_);
}
